// RIConv2FeaturePropagation_v2_5282809774329
// MI455X (gfx1250) — matrix operand order repaired and checked by reading, not compiled, not run on hardware
//
#include <hip/hip_runtime.h>
#include <math.h>

typedef _Float16 v16h __attribute__((ext_vector_type(16)));
typedef float    v8f  __attribute__((ext_vector_type(8)));
typedef _Float16 h2v  __attribute__((ext_vector_type(2)));

#define KNN 8
#define EPSRI 1e-7f
#define BNEPS 1e-5f

struct F3 { float x, y, z; };
__device__ __forceinline__ F3 mk3(const float* p){ return {p[0], p[1], p[2]}; }
__device__ __forceinline__ F3 sub3(F3 a, F3 b){ return {a.x-b.x, a.y-b.y, a.z-b.z}; }
__device__ __forceinline__ float dot3(F3 a, F3 b){ return a.x*b.x + a.y*b.y + a.z*b.z; }
__device__ __forceinline__ F3 scale3(F3 a, float s){ return {a.x*s, a.y*s, a.z*s}; }
__device__ __forceinline__ F3 cross3(F3 a, F3 b){
  return {a.y*b.z - a.z*b.y, a.z*b.x - a.x*b.z, a.x*b.y - a.y*b.x};
}
__device__ __forceinline__ float signf_(float v){ return (v > 0.f) ? 1.f : ((v < 0.f) ? -1.f : 0.f); }

// ---------------------------------------------------------------------------
// 1) kNN: one thread per query point, serial scan over S=1024, sorted top-8.
// ---------------------------------------------------------------------------
__global__ __launch_bounds__(256) void knn_kernel(
    const float* __restrict__ xyz1, const float* __restrict__ xyz2,
    int* __restrict__ idx_out, int B, int N, int S)
{
  int gid = blockIdx.x * blockDim.x + threadIdx.x;
  if (gid >= B * N) return;
  int b = gid / N;
  const float* p1 = xyz1 + (size_t)gid * 3;
  float px = p1[0], py = p1[1], pz = p1[2];
  const float* q = xyz2 + (size_t)b * S * 3;
  float bd[KNN]; int bi[KNN];
  #pragma unroll
  for (int k = 0; k < KNN; ++k) { bd[k] = 3.0e38f; bi[k] = 0; }
  for (int s = 0; s < S; ++s) {
    float dx = q[3*s+0] - px, dy = q[3*s+1] - py, dz = q[3*s+2] - pz;
    float d = dx*dx + dy*dy + dz*dz;
    if (d < bd[KNN-1]) {
      int pos = KNN - 1;
      while (pos > 0 && d < bd[pos-1]) { bd[pos] = bd[pos-1]; bi[pos] = bi[pos-1]; --pos; }
      bd[pos] = d; bi[pos] = s;
    }
  }
  #pragma unroll
  for (int k = 0; k < KNN; ++k) idx_out[(size_t)gid * KNN + k] = bi[k];
}

// ---------------------------------------------------------------------------
// 2) Rotation-invariant features; writes 8 feature rows of Xri [32 x M1] f16
//    and reorders idx in place.
// ---------------------------------------------------------------------------
__global__ __launch_bounds__(256) void ri_kernel(
    const float* __restrict__ xyz1, const float* __restrict__ xyz2,
    const float* __restrict__ norm1, const float* __restrict__ norm2,
    int* __restrict__ idx_io, _Float16* __restrict__ Xri,
    int B, int N, int S, int M1)
{
  int gid = blockIdx.x * blockDim.x + threadIdx.x;
  if (gid >= B * N) return;
  int b = gid / N;
  F3 p  = mk3(xyz1  + (size_t)gid * 3);
  F3 nn = mk3(norm1 + (size_t)gid * 3);
  const float* x2 = xyz2  + (size_t)b * S * 3;
  const float* n2 = norm2 + (size_t)b * S * 3;
  size_t m1base = (size_t)gid * KNN;

  int idx[KNN];
  #pragma unroll
  for (int k = 0; k < KNN; ++k) idx[k] = idx_io[m1base + k];

  F3 punit[KNN]; float plen[KNN];
  #pragma unroll
  for (int k = 0; k < KNN; ++k) {
    F3 g = sub3(mk3(x2 + 3*idx[k]), p);
    float dp = dot3(g, nn);
    F3 proj = sub3(g, scale3(nn, dp));
    float L = sqrtf(dot3(proj, proj));
    plen[k] = L;
    punit[k] = (L > 0.f) ? scale3(proj, 1.f / L) : F3{0.f, 0.f, 0.f};
  }
  int mi = 0; float mv = plen[0];
  #pragma unroll
  for (int k = 1; k < KNN; ++k) if (plen[k] > mv) { mv = plen[k]; mi = k; }
  F3 vref = punit[mi];

  float dots[KNN];
  #pragma unroll
  for (int k = 0; k < KNN; ++k) {
    float d  = dot3(punit[k], vref);
    float sg = (k == 0) ? 1.f : signf_(dot3(cross3(punit[k], vref), nn));
    dots[k] = sg * d - (1.f - sg);
  }
  // stable descending argsort of dots
  int order[KNN];
  #pragma unroll
  for (int k = 0; k < KNN; ++k) order[k] = k;
  for (int i = 1; i < KNN; ++i) {
    int key = order[i]; float kv = dots[key]; int j = i - 1;
    while (j >= 0 && dots[order[j]] < kv) { order[j+1] = order[j]; --j; }
    order[j+1] = key;
  }
  float dsrt[KNN]; int io[KNN];
  #pragma unroll
  for (int k = 0; k < KNN; ++k) { io[k] = idx[order[k]]; dsrt[k] = dots[order[k]]; }

  F3 g2[KNN], gu[KNN], gn[KNN]; float glen[KNN];
  #pragma unroll
  for (int k = 0; k < KNN; ++k) {
    F3 g = sub3(mk3(x2 + 3*io[k]), p);
    float L = sqrtf(dot3(g, g));
    g2[k] = g; glen[k] = L;
    gu[k] = (L > 0.f) ? scale3(g, 1.f / L) : F3{0.f, 0.f, 0.f};
    gn[k] = mk3(n2 + 3*io[k]);
  }
  #pragma unroll
  for (int k = 0; k < KNN; ++k) {
    int km = (k + KNN - 1) % KNN;
    float a0 = dot3(gu[k], nn);
    float a1 = dot3(gu[k], gn[k]);
    float cv = fminf(fmaxf(dot3(gn[k], nn), -1.f + EPSRI), 1.f - EPSRI);
    float an = acosf(cv); an = (a0 < a1) ? an : -an;
    F3 iv = sub3(g2[k], g2[km]);
    float L = sqrtf(dot3(iv, iv));
    F3 iu = (L > 0.f) ? scale3(iv, 1.f / L) : F3{0.f, 0.f, 0.f};
    float ia0 = dot3(iu, gn[k]);
    float ia1 = dot3(iu, gn[km]);
    float cv2 = fminf(fmaxf(dot3(gn[k], gn[km]), -1.f + EPSRI), 1.f - EPSRI);
    float ia2 = acosf(cv2); ia2 = (ia0 < ia1) ? ia2 : -ia2;
    float pf = (k == 0) ? (-3.f - dsrt[KNN-1]) : (dsrt[k] - dsrt[k-1]);
    size_t col = m1base + k;
    Xri[(size_t)0*M1 + col] = (_Float16)glen[k];
    Xri[(size_t)1*M1 + col] = (_Float16)pf;
    Xri[(size_t)2*M1 + col] = (_Float16)a0;
    Xri[(size_t)3*M1 + col] = (_Float16)a1;
    Xri[(size_t)4*M1 + col] = (_Float16)an;
    Xri[(size_t)5*M1 + col] = (_Float16)ia0;
    Xri[(size_t)6*M1 + col] = (_Float16)ia1;
    Xri[(size_t)7*M1 + col] = (_Float16)ia2;
    idx_io[col] = io[k];
  }
}

// ---------------------------------------------------------------------------
// Utility kernels
// ---------------------------------------------------------------------------
__global__ __launch_bounds__(256) void fill_h_kernel(_Float16* p, long n) {
  long i = (long)blockIdx.x * blockDim.x + threadIdx.x;
  if (i < n) p[i] = (_Float16)0.f;
}

__global__ __launch_bounds__(256) void wconv_kernel(
    const float* __restrict__ W, _Float16* __restrict__ Wh,
    int Cout, int Cin, int CinPad)
{
  int i = blockIdx.x * blockDim.x + threadIdx.x;
  if (i >= Cout * CinPad) return;
  int r = i / CinPad, c = i % CinPad;
  Wh[i] = (c < Cin) ? (_Float16)W[(size_t)r * Cin + c] : (_Float16)0.f;
}

__global__ __launch_bounds__(256) void gather_kernel(
    const float* __restrict__ points2, const int* __restrict__ idx,
    _Float16* __restrict__ Xcat, int C2, int M1, int NK, int S)
{
  long i = (long)blockIdx.x * blockDim.x + threadIdx.x;
  if (i >= (long)C2 * M1) return;
  int ch = (int)(i / M1); int m1 = (int)(i % M1);
  int b = m1 / NK;
  int j = idx[m1];
  float v = points2[((size_t)b * C2 + ch) * S + j];
  Xcat[(size_t)(64 + ch) * M1 + m1] = (_Float16)v;
}

__global__ __launch_bounds__(256) void copyp1_kernel(
    const float* __restrict__ points1, _Float16* __restrict__ Xc2,
    int C1, int M2, int N)
{
  long i = (long)blockIdx.x * blockDim.x + threadIdx.x;
  if (i >= (long)C1 * M2) return;
  int ch = (int)(i / M2); int m2 = (int)(i % M2);
  int b = m2 / N, n = m2 % N;
  Xc2[(size_t)(256 + ch) * M2 + m2] = (_Float16)points1[((size_t)b * C1 + ch) * N + n];
}

__global__ __launch_bounds__(256) void maxk_kernel(
    const _Float16* __restrict__ Xin, _Float16* __restrict__ Xout,
    int C, int M2)
{
  long i = (long)blockIdx.x * blockDim.x + threadIdx.x;
  if (i >= (long)C * M2) return;
  int ch = (int)(i / M2); int m2 = (int)(i % M2);
  const _Float16* p = Xin + ((size_t)ch * M2 + m2) * KNN;
  float m = (float)p[0];
  #pragma unroll
  for (int k = 1; k < KNN; ++k) m = fmaxf(m, (float)p[k]);
  Xout[(size_t)ch * M2 + m2] = (_Float16)m;
}

__global__ __launch_bounds__(256) void stats_kernel(
    const _Float16* __restrict__ Y, float* __restrict__ mean,
    float* __restrict__ var, int M)
{
  int ch = blockIdx.x;
  const _Float16* row = Y + (size_t)ch * M;
  float s = 0.f, s2 = 0.f;
  for (int i = threadIdx.x; i < M; i += 256) {
    float v = (float)row[i]; s += v; s2 += v * v;
  }
  __shared__ float bs[256], bs2[256];
  bs[threadIdx.x] = s; bs2[threadIdx.x] = s2;
  __syncthreads();
  for (int off = 128; off > 0; off >>= 1) {
    if (threadIdx.x < off) { bs[threadIdx.x] += bs[threadIdx.x+off]; bs2[threadIdx.x] += bs2[threadIdx.x+off]; }
    __syncthreads();
  }
  if (threadIdx.x == 0) {
    float m = bs[0] / M;
    mean[ch] = m;
    var[ch] = bs2[0] / M - m * m;
  }
}

__global__ __launch_bounds__(256) void bnrelu_kernel(
    _Float16* __restrict__ Y, const float* __restrict__ gamma,
    const float* __restrict__ beta, const float* __restrict__ mean,
    const float* __restrict__ var, int M, long total)
{
  long i = (long)blockIdx.x * blockDim.x + threadIdx.x;
  if (i >= total) return;
  int ch = (int)(i / M);
  float x = (float)Y[i];
  float y = gamma[ch] * (x - mean[ch]) * rsqrtf(var[ch] + BNEPS) + beta[ch];
  Y[i] = (_Float16)fmaxf(y, 0.f);
}

__global__ __launch_bounds__(256) void bnrelu_final_kernel(
    const _Float16* __restrict__ Y, const float* __restrict__ gamma,
    const float* __restrict__ beta, const float* __restrict__ mean,
    const float* __restrict__ var, float* __restrict__ out,
    int M, int N, int Cout)
{
  long i = (long)blockIdx.x * blockDim.x + threadIdx.x;
  if (i >= (long)Cout * M) return;
  int ch = (int)(i / M); int m = (int)(i % M);
  int b = m / N, n = m % N;
  float x = (float)Y[i];
  float y = gamma[ch] * (x - mean[ch]) * rsqrtf(var[ch] + BNEPS) + beta[ch];
  out[((size_t)b * Cout + ch) * N + n] = fmaxf(y, 0.f);
}

// ---------------------------------------------------------------------------
// WMMA GEMM: Y[Cout x M] = (W[Cout x Kdim] @ X[Kdim x M]) + bias.
// Block = 256 threads = 8 waves; block tile = 32 rows x 128 cols; each wave
// owns TWO 16x16 f32 accumulators (rows 0-15 / 16-31) sharing one B fragment,
// issuing two independent v_wmma_f32_16x16x32_f16 per K-step.
// The 32-row A slab (contiguous in memory) is async-DMA'd to LDS once via
// global_load_async_to_lds_b128 / s_wait_asynccnt; the X tile is staged
// per K-step with b128 global loads + ds transpose stores.
// Requires: Cout%32==0, Kdim%32==0 (<=384), M%128==0.
// ---------------------------------------------------------------------------
__global__ __launch_bounds__(256) void gemm_wmma_kernel(
    const _Float16* __restrict__ A, const _Float16* __restrict__ X,
    const float* __restrict__ bias, _Float16* __restrict__ Y,
    int Cout, int Kdim, int M)
{
  __shared__ _Float16 sA[32 * 384];      // [row][k], full K slab for 32 rows
  __shared__ _Float16 sXt[128 * 32];     // [col][k] transposed stage
  int t = threadIdx.x;
  int wave = t >> 5, lane = t & 31;
  int rowBase = blockIdx.y * 32;
  int colBase = blockIdx.x * 128;
  int colLocal = wave * 16 + (lane & 15);
  int hi = lane >> 4;
  int r = lane & 15;
  v8f acc0 = {}, acc1 = {};

  // --- Async copy of A slab (32 rows x Kdim halves, contiguous) to LDS. ---
  {
    int slabBytes = 32 * Kdim * 2;
    const char* gsrc = (const char*)(A + (size_t)rowBase * Kdim);
    unsigned lbase = (unsigned)(uintptr_t)&sA[0];
    for (int ofs = t * 16; ofs < slabBytes; ofs += 256 * 16) {
      unsigned long long ga = (unsigned long long)(uintptr_t)(gsrc + ofs);
      unsigned la = lbase + (unsigned)ofs;
      asm volatile("global_load_async_to_lds_b128 %0, %1, off"
                   :: "v"(la), "v"(ga) : "memory");
    }
    asm volatile("s_wait_asynccnt 0x0" ::: "memory");
  }
  __syncthreads();

  for (int k0 = 0; k0 < Kdim; k0 += 32) {
    // Stage X tile transposed: 2 x 256 threads x 8 halves.
    #pragma unroll
    for (int i = 0; i < 2; ++i) {
      int idx = t + i * 256;
      int kr = idx >> 4;     // k row within tile: 0..31
      int ch = idx & 15;     // 8-col chunk
      const _Float16* src = X + (size_t)(k0 + kr) * M + colBase + ch * 8;
      uint4 d = *(const uint4*)src;
      _Float16 tmp[8]; *(uint4*)tmp = d;
      #pragma unroll
      for (int j = 0; j < 8; ++j) sXt[(ch * 8 + j) * 32 + kr] = tmp[j];
      if (k0 + 32 < Kdim) __builtin_prefetch(src + (size_t)32 * M, 0, 1);
    }
    __syncthreads();

    // Build B fragment (32x16 f16): lane = col (l&15), K = 8*(l>>4)+s (s<8), 16+8*(l>>4)+s-8 (s>=8).
    v16h bf;
    #pragma unroll
    for (int v = 0; v < 8; ++v) {
      h2v p = *(const h2v*)(&sXt[colLocal * 32 + hi * 8 + (v >> 2) * 16 + 2 * (v & 3)]);
      bf[2*v] = p[0]; bf[2*v+1] = p[1];
    }
    // Build the two A fragments (16x32 f16 each) from the staged slab.
    const _Float16* aRow0 = &sA[(size_t)r * Kdim + k0];
    const _Float16* aRow1 = &sA[(size_t)(16 + r) * Kdim + k0];
    v16h af0, af1;
    #pragma unroll
    for (int v = 0; v < 4; ++v) {
      h2v p0 = *(const h2v*)(aRow0 + hi * 8 + 2 * v);
      af0[2*v]   = p0[0]; af0[2*v+1] = p0[1];
      h2v q0 = *(const h2v*)(aRow0 + 16 + hi * 8 + 2 * v);
      af0[8+2*v] = q0[0]; af0[9+2*v] = q0[1];
      h2v p1 = *(const h2v*)(aRow1 + hi * 8 + 2 * v);
      af1[2*v]   = p1[0]; af1[2*v+1] = p1[1];
      h2v q1 = *(const h2v*)(aRow1 + 16 + hi * 8 + 2 * v);
      af1[8+2*v] = q1[0]; af1[9+2*v] = q1[1];
    }
    // Two independent WMMAs sharing bf -> back-to-back XDL issue.
    acc0 = __builtin_amdgcn_wmma_f32_16x16x32_f16(
        false, af0, false, bf, (short)0, acc0, false, false);
    acc1 = __builtin_amdgcn_wmma_f32_16x16x32_f16(
        false, af1, false, bf, (short)0, acc1, false, false);
    __syncthreads();
  }

  // D layout: col = lane&15, row = v + 8*(lane>>4). Add bias, store f16.
  int col = colBase + colLocal;
  #pragma unroll
  for (int v = 0; v < 8; ++v) {
    int ch0 = rowBase + v + 8 * hi;
    Y[(size_t)ch0 * M + col] = (_Float16)(acc0[v] + bias[ch0]);
    int ch1 = rowBase + 16 + v + 8 * hi;
    Y[(size_t)ch1 * M + col] = (_Float16)(acc1[v] + bias[ch1]);
  }
}

// ---------------------------------------------------------------------------
// Host launch
// ---------------------------------------------------------------------------
extern "C" void kernel_launch(void* const* d_in, const int* in_sizes, int n_in,
                              void* d_out, int out_size, void* d_ws, size_t ws_size,
                              hipStream_t stream) {
  (void)in_sizes; (void)n_in; (void)out_size; (void)ws_size;
  const int B = 4, N = 4096, S = 1024, C1 = 128, C2 = 256;
  const int M1 = B * N * KNN;   // 131072
  const int M2 = B * N;         // 16384
  const int NP = B * N;

  const float* xyz1    = (const float*)d_in[0];
  const float* xyz2    = (const float*)d_in[1];
  const float* norm1   = (const float*)d_in[2];
  const float* norm2   = (const float*)d_in[3];
  const float* points1 = (const float*)d_in[4];
  const float* points2 = (const float*)d_in[5];
  const float* Wl[6];  const float* bl[6];  const float* gl[6];  const float* el[6];
  for (int L = 0; L < 6; ++L) {
    Wl[L] = (const float*)d_in[6 + 4*L + 0];
    bl[L] = (const float*)d_in[6 + 4*L + 1];
    gl[L] = (const float*)d_in[6 + 4*L + 2];
    el[L] = (const float*)d_in[6 + 4*L + 3];
  }
  const int CoutL[6] = {32, 64, 256, 256, 256, 128};
  const int CinL[6]  = {8, 32, 320, 256, 384, 256};
  const int KpadL[6] = {32, 32, 320, 256, 384, 256};

  char* ws = (char*)d_ws;
  size_t off = 0;
  auto carve = [&](size_t bytes) -> char* {
    char* p = ws + off;
    off = (off + bytes + 255) & ~(size_t)255;
    return p;
  };
  int*      idxb = (int*)     carve((size_t)M1 * 4);
  _Float16* Xri  = (_Float16*)carve((size_t)32  * M1 * 2);
  _Float16* Xh1  = (_Float16*)carve((size_t)32  * M1 * 2);
  _Float16* Xcat = (_Float16*)carve((size_t)320 * M1 * 2);
  _Float16* Xm1  = (_Float16*)carve((size_t)256 * M1 * 2);
  _Float16* Xm2  = (_Float16*)carve((size_t)256 * M1 * 2);
  _Float16* Xc2  = (_Float16*)carve((size_t)384 * M2 * 2);
  _Float16* Xo1  = (_Float16*)carve((size_t)256 * M2 * 2);
  _Float16* Xo2  = (_Float16*)carve((size_t)128 * M2 * 2);
  _Float16* Wh[6];
  for (int L = 0; L < 6; ++L)
    Wh[L] = (_Float16*)carve((size_t)CoutL[L] * KpadL[L] * 2);
  float* st = (float*)carve((size_t)6 * 512 * 4);

  // Geometry
  knn_kernel<<<(NP + 255) / 256, 256, 0, stream>>>(xyz1, xyz2, idxb, B, N, S);
  ri_kernel <<<(NP + 255) / 256, 256, 0, stream>>>(xyz1, xyz2, norm1, norm2,
                                                   idxb, Xri, B, N, S, M1);
  {
    long n = (long)24 * M1;   // zero f16 padding rows 8..31 of Xri
    fill_h_kernel<<<(int)((n + 255) / 256), 256, 0, stream>>>(Xri + (size_t)8 * M1, n);
  }
  for (int L = 0; L < 6; ++L) {
    int tot = CoutL[L] * KpadL[L];
    wconv_kernel<<<(tot + 255) / 256, 256, 0, stream>>>(Wl[L], Wh[L],
                                                        CoutL[L], CinL[L], KpadL[L]);
  }

  auto run_layer = [&](int L, const _Float16* Xin, _Float16* Yout, int M) {
    int Cout = CoutL[L], Kd = KpadL[L];
    dim3 grid(M / 128, Cout / 32);
    gemm_wmma_kernel<<<grid, 256, 0, stream>>>(Wh[L], Xin, bl[L], Yout, Cout, Kd, M);
    float* mean = st + (size_t)L * 512;
    float* var  = mean + 256;
    stats_kernel<<<Cout, 256, 0, stream>>>(Yout, mean, var, M);
    long tot = (long)Cout * M;
    bnrelu_kernel<<<(int)((tot + 255) / 256), 256, 0, stream>>>(Yout, gl[L], el[L],
                                                                mean, var, M, tot);
  };

  run_layer(0, Xri, Xh1, M1);        // 8(->32) -> 32
  run_layer(1, Xh1, Xcat, M1);       // 32 -> 64 written into Xcat rows 0..63
  {
    long tot = (long)C2 * M1;
    gather_kernel<<<(int)((tot + 255) / 256), 256, 0, stream>>>(points2, idxb, Xcat,
                                                                C2, M1, N * KNN, S);
  }
  run_layer(2, Xcat, Xm1, M1);       // 320 -> 256
  run_layer(3, Xm1, Xm2, M1);        // 256 -> 256
  {
    long tot = (long)256 * M2;
    maxk_kernel<<<(int)((tot + 255) / 256), 256, 0, stream>>>(Xm2, Xc2, 256, M2);
    long tot2 = (long)C1 * M2;
    copyp1_kernel<<<(int)((tot2 + 255) / 256), 256, 0, stream>>>(points1, Xc2, C1, M2, N);
  }
  run_layer(4, Xc2, Xo1, M2);        // 384 -> 256

  {
    int L = 5, Cout = CoutL[L], Kd = KpadL[L];
    dim3 grid(M2 / 128, Cout / 32);
    gemm_wmma_kernel<<<grid, 256, 0, stream>>>(Wh[L], Xo1, bl[L], Xo2, Cout, Kd, M2);
    float* mean = st + (size_t)L * 512;
    float* var  = mean + 256;
    stats_kernel<<<Cout, 256, 0, stream>>>(Xo2, mean, var, M2);
    long tot = (long)Cout * M2;
    bnrelu_final_kernel<<<(int)((tot + 255) / 256), 256, 0, stream>>>(
        Xo2, gl[L], el[L], mean, var, (float*)d_out, M2, N, Cout);
  }
}
